// GIN_40802189312202
// MI455X (gfx1250) — compile-verified, matrix operand order checked by reading, not run on hardware
//
#include <hip/hip_runtime.h>

#define NNODES 100000
#define NEDGES 1600000
#define NGRAPHS 512
#define DH 128
#define DOUT 10

typedef __attribute__((ext_vector_type(16))) __bf16 v16bf;
typedef __attribute__((ext_vector_type(8)))  float  v8f;
typedef unsigned short u16t;
typedef __attribute__((ext_vector_type(8))) u16t u16x8;
typedef __attribute__((ext_vector_type(4))) u16t u16x4;

union Frag16 {
  v16bf v;
  u16x8 h[2];
  u16t  u[16];
};

__device__ __forceinline__ u16t f32_to_bf16(float f) {
  unsigned int u = __float_as_uint(f);
  unsigned int r = u + 0x7FFFu + ((u >> 16) & 1u);   // round-to-nearest-even
  return (u16t)(r >> 16);
}

// ---------------------------------------------------------------------------
// Y[M x 128] = act( A_bf16[M x 128] @ W_bf16[128 x 128] + bias )
// One wave per 16x16 output tile; wave's column strip fixed so B fragments for
// all 4 K-steps are loaded once and reused across a grid-stride row-tile loop.
// ---------------------------------------------------------------------------
template <bool RELU, bool OUT_BF16>
__global__ __launch_bounds__(256)
void gin_gemm_wmma(const u16t* __restrict__ A, const u16t* __restrict__ W,
                   const float* __restrict__ bias,
                   float* __restrict__ outF, u16t* __restrict__ outB, int M) {
  const int lane    = threadIdx.x & 31;
  const int wave    = blockIdx.x * (blockDim.x >> 5) + (threadIdx.x >> 5);
  const int nWaves  = gridDim.x * (blockDim.x >> 5);
  const int colTile = wave & 7;                 // N=128 -> 8 column tiles
  const int rt0     = wave >> 3;
  const int rtStep  = nWaves >> 3;
  const int nRT     = M >> 4;

  const int colBase = colTile * 16;
  const int halfSel = (lane >> 4) & 1;          // lanes 0-15 vs 16-31
  const int rc      = lane & 15;                // A-row / B-col / C-col within tile

  // Preload B fragments for all 4 K-steps (ISA 16-bit A/B lane layout:
  // lane half 0 holds K = kb..kb+7 and kb+16..kb+23 with kb = ks*32;
  // lane half 1 the same shifted by 8).
  Frag16 bf[4];
#pragma unroll
  for (int ks = 0; ks < 4; ++ks) {
    const int kb = ks * 32 + halfSel * 8;
#pragma unroll
    for (int v = 0; v < 8; ++v) {
      bf[ks].u[v]     = W[(kb + v) * DH      + colBase + rc];
      bf[ks].u[v + 8] = W[(kb + 16 + v) * DH + colBase + rc];
    }
  }
  const float bv = bias[colBase + rc];

  for (int rt = rt0; rt < nRT; rt += rtStep) {
    const int rowBase = rt * 16;
    const u16t* arow = A + (size_t)(rowBase + rc) * DH;
    v8f acc = {};
#pragma unroll
    for (int ks = 0; ks < 4; ++ks) {
      const int kb = ks * 32 + halfSel * 8;
      Frag16 af;
      af.h[0] = *(const u16x8*)(arow + kb);        // K = kb..kb+7
      af.h[1] = *(const u16x8*)(arow + kb + 16);   // K = kb+16..kb+23
      acc = __builtin_amdgcn_wmma_f32_16x16x32_bf16(
          false, af.v, false, bf[ks].v, (short)0, acc, false, false);
    }
    // C/D layout: VGPR i -> M = i (lanes 0-15) or i+8 (lanes 16-31), N = rc.
#pragma unroll
    for (int i = 0; i < 8; ++i) {
      const int row = rowBase + i + halfSel * 8;
      float val = acc[i] + bv;
      if (RELU) val = fmaxf(val, 0.0f);
      if (OUT_BF16) outB[(size_t)row * DH + colBase + rc] = f32_to_bf16(val);
      else          outF[(size_t)row * DH + colBase + rc] = val;
    }
  }
}

// agg[dst] += src_row ; one wave per edge, 4 floats per lane.
__global__ __launch_bounds__(256)
void gin_scatter_add(const float* __restrict__ xin, const int* __restrict__ src,
                     const int* __restrict__ dst, float* __restrict__ agg) {
  const int wid  = (blockIdx.x * blockDim.x + threadIdx.x) >> 5;
  const int lane = threadIdx.x & 31;
  if (wid >= NEDGES) return;
  const int s = src[wid];
  const int d = dst[wid];
  const float4 v = *(const float4*)(xin + (size_t)s * DH + lane * 4);
  float* p = agg + (size_t)d * DH + lane * 4;
  unsafeAtomicAdd(p + 0, v.x);
  unsafeAtomicAdd(p + 1, v.y);
  unsafeAtomicAdd(p + 2, v.z);
  unsafeAtomicAdd(p + 3, v.w);
}

__global__ __launch_bounds__(256)
void gin_copy4(const float* __restrict__ in, float* __restrict__ out, int n4) {
  const int i = blockIdx.x * blockDim.x + threadIdx.x;
  if (i < n4) ((float4*)out)[i] = ((const float4*)in)[i];
}

__global__ __launch_bounds__(256)
void gin_cvt_bf16(const float* __restrict__ in, u16t* __restrict__ out, int n4) {
  const int i = blockIdx.x * blockDim.x + threadIdx.x;
  if (i >= n4) return;
  const float4 v = ((const float4*)in)[i];
  u16x4 o;
  o.x = f32_to_bf16(v.x); o.y = f32_to_bf16(v.y);
  o.z = f32_to_bf16(v.z); o.w = f32_to_bf16(v.w);
  ((u16x4*)out)[i] = o;
}

__global__ __launch_bounds__(256)
void gin_zero4(float* __restrict__ p, int n4) {
  const int i = blockIdx.x * blockDim.x + threadIdx.x;
  if (i < n4) ((float4*)p)[i] = make_float4(0.f, 0.f, 0.f, 0.f);
}

// g[batch[node]] += h[node] ; 32 lanes x float4 per node.
__global__ __launch_bounds__(256)
void gin_pool_add(const float* __restrict__ h, const int* __restrict__ batch,
                  float* __restrict__ g) {
  const int t    = blockIdx.x * blockDim.x + threadIdx.x;
  const int node = t >> 5;
  const int lane = t & 31;
  if (node >= NNODES) return;
  const int b = batch[node];
  const float4 v = *(const float4*)(h + (size_t)node * DH + lane * 4);
  float* p = g + (size_t)b * DH + lane * 4;
  unsafeAtomicAdd(p + 0, v.x);
  unsafeAtomicAdd(p + 1, v.y);
  unsafeAtomicAdd(p + 2, v.z);
  unsafeAtomicAdd(p + 3, v.w);
}

// out[g][o] = sum_k t2[g][k] * w[k][o] + b[o]   (512 x 10, trivial)
__global__ __launch_bounds__(64)
void gin_classifier(const float* __restrict__ t2, const float* __restrict__ w,
                    const float* __restrict__ b, float* __restrict__ out) {
  const int i = blockIdx.x * blockDim.x + threadIdx.x;
  if (i >= NGRAPHS * DOUT) return;
  const int gi = i / DOUT;
  const int o  = i % DOUT;
  float acc = b[o];
  const float* row = t2 + (size_t)gi * DH;
#pragma unroll 4
  for (int k = 0; k < DH; ++k) acc = fmaf(row[k], w[k * DOUT + o], acc);
  out[(size_t)gi * DOUT + o] = acc;
}

extern "C" void kernel_launch(void* const* d_in, const int* in_sizes, int n_in,
                              void* d_out, int out_size, void* d_ws, size_t ws_size,
                              hipStream_t stream) {
  const float* x     = (const float*)d_in[0];
  const int*   eidx  = (const int*)d_in[1];
  const int*   src   = eidx;
  const int*   dst   = eidx + NEDGES;
  const int*   batch = (const int*)d_in[2];
  const float* c0w1 = (const float*)d_in[3];  const float* c0b1 = (const float*)d_in[4];
  const float* c0w2 = (const float*)d_in[5];  const float* c0b2 = (const float*)d_in[6];
  const float* c1w1 = (const float*)d_in[7];  const float* c1b1 = (const float*)d_in[8];
  const float* c1w2 = (const float*)d_in[9];  const float* c1b2 = (const float*)d_in[10];
  const float* mw1  = (const float*)d_in[11]; const float* mb1  = (const float*)d_in[12];
  const float* mw2  = (const float*)d_in[13]; const float* mb2  = (const float*)d_in[14];
  const float* clsw = (const float*)d_in[15]; const float* clsb = (const float*)d_in[16];
  float* out = (float*)d_out;

  // Workspace layout
  char* ws = (char*)d_ws;
  size_t off = 0;
  float* aggF = (float*)(ws + off); off += (size_t)NNODES * DH * 4;   // 51.2 MB
  float* hF   = (float*)(ws + off); off += (size_t)NNODES * DH * 4;   // 51.2 MB
  u16t*  aBF  = (u16t*)(ws + off);  off += (size_t)NNODES * DH * 2;   // 25.6 MB
  u16t*  tBF  = (u16t*)(ws + off);  off += (size_t)NNODES * DH * 2;   // 25.6 MB
  u16t*  wB[6];
  for (int i = 0; i < 6; ++i) { wB[i] = (u16t*)(ws + off); off += (size_t)DH * DH * 2; }
  float* gF  = (float*)(ws + off);  off += (size_t)NGRAPHS * DH * 4;
  u16t*  gBF = (u16t*)(ws + off);   off += (size_t)NGRAPHS * DH * 2;

  const int nodeV4  = NNODES * DH / 4;         // 3.2M
  const int wV4     = DH * DH / 4;             // 4096
  const int gV4     = NGRAPHS * DH / 4;        // 16384
  const int CPB     = 256;
  const int gemmBlk = 640;                      // 5120 waves -> ~10 row tiles each

  // 1) weights -> bf16
  const float* wsrc[6] = {c0w1, c0w2, c1w1, c1w2, mw1, mw2};
  for (int i = 0; i < 6; ++i)
    gin_cvt_bf16<<<(wV4 + CPB - 1) / CPB, CPB, 0, stream>>>(wsrc[i], wB[i], wV4);

  // 2) conv0: agg = x + sum_neighbors(x)
  gin_copy4<<<nodeV4 / CPB, CPB, 0, stream>>>(x, aggF, nodeV4);
  gin_scatter_add<<<NEDGES / 8, CPB, 0, stream>>>(x, src, dst, aggF);
  gin_cvt_bf16<<<nodeV4 / CPB, CPB, 0, stream>>>(aggF, aBF, nodeV4);
  gin_gemm_wmma<true, true ><<<gemmBlk, CPB, 0, stream>>>(aBF, wB[0], c0b1, nullptr, tBF, NNODES);
  gin_gemm_wmma<true, false><<<gemmBlk, CPB, 0, stream>>>(tBF, wB[1], c0b2, hF, nullptr, NNODES);

  // 3) conv1: agg = h + sum_neighbors(h)
  gin_copy4<<<nodeV4 / CPB, CPB, 0, stream>>>(hF, aggF, nodeV4);
  gin_scatter_add<<<NEDGES / 8, CPB, 0, stream>>>(hF, src, dst, aggF);
  gin_cvt_bf16<<<nodeV4 / CPB, CPB, 0, stream>>>(aggF, aBF, nodeV4);
  gin_gemm_wmma<true, true ><<<gemmBlk, CPB, 0, stream>>>(aBF, wB[2], c1b1, nullptr, tBF, NNODES);
  gin_gemm_wmma<true, false><<<gemmBlk, CPB, 0, stream>>>(tBF, wB[3], c1b2, hF, nullptr, NNODES);

  // 4) global add pool -> [512,128]
  gin_zero4<<<(gV4 + CPB - 1) / CPB, CPB, 0, stream>>>(gF, gV4);
  gin_pool_add<<<(NNODES * 32) / CPB, CPB, 0, stream>>>(hF, batch, gF);

  // 5) post-pool MLP (M=512) through the same WMMA path
  gin_cvt_bf16<<<(gV4 + CPB - 1) / CPB, CPB, 0, stream>>>(gF, gBF, gV4);
  gin_gemm_wmma<true, true ><<<32, CPB, 0, stream>>>(gBF, wB[4], mb1, nullptr, tBF, NGRAPHS);
  gin_gemm_wmma<true, false><<<32, CPB, 0, stream>>>(tBF, wB[5], mb2, gF, nullptr, NGRAPHS);

  // 6) classifier [512,10]
  gin_classifier<<<(NGRAPHS * DOUT + 63) / 64, 64, 0, stream>>>(gF, clsw, clsb, out);

  (void)in_sizes; (void)n_in; (void)out_size; (void)ws_size;
}
